// VAE_31482110279909
// MI455X (gfx1250) — matrix operand order repaired and checked by reading, not compiled, not run on hardware
//
#include <hip/hip_runtime.h>
#include <hip/hip_bf16.h>

// ---------------------------------------------------------------------------
// CDNA5 wave32 WMMA helpers (v_wmma_f32_16x16x32_f16: D = A(16x32) B(32x16) + C)
// ---------------------------------------------------------------------------
typedef _Float16 h16_t;
typedef __attribute__((ext_vector_type(16))) _Float16 v16h;
typedef __attribute__((ext_vector_type(8)))  _Float16 v8h;
typedef __attribute__((ext_vector_type(4)))  _Float16 v4h;
typedef __attribute__((ext_vector_type(8)))  float    v8f;
typedef __attribute__((ext_vector_type(4)))  float    v4f;

// Global (addrspace 1) views so weight traffic lowers to global_load_b128
// (FLAT would tick both LOADcnt and DScnt and burn 64-bit VGPR address math).
typedef __attribute__((address_space(1))) const h16_t ghalf;
typedef __attribute__((address_space(1))) const v8h   gv8h;

union V16U { v16h v; v8h h[2]; };

__device__ __forceinline__ int lane_id() { return (int)(threadIdx.x & 31u); }

// Defeat loop-invariant hoisting (prevents preloading all weight tiles and
// spilling them to scratch across the sequential scans), then re-assert the
// global address space so loads stay global_load_* with SGPR base.
__device__ __forceinline__ const ghalf* launder_g(const h16_t* p) {
  unsigned long long v = (unsigned long long)p;
  asm volatile("" : "+s"(v));
  return (const ghalf*)v;
}
// Plain generic->global cast (no anti-hoist) for non-loop kernels.
__device__ __forceinline__ const ghalf* to_g(const h16_t* p) {
  return (const ghalf*)(unsigned long long)p;
}

// A tile (16 rows x 32 K) from row-major f16, stride ld (multiple of 8).
// Lane L: M = L&15, holds K = 8*(L>>4) + {0..7} and +16 (ISA 7.12.2 A layout).
__device__ __forceinline__ v16h load_a_f16(const h16_t* base, int ld) {
  const int l = lane_id();
  const h16_t* p = base + (l & 15) * ld + ((l >> 4) << 3);
  V16U u;
  u.h[0] = *reinterpret_cast<const v8h*>(p);
  u.h[1] = *reinterpret_cast<const v8h*>(p + 16);
  return u.v;
}
__device__ __forceinline__ v16h load_a_f16(const ghalf* base, int ld) {
  const int l = lane_id();
  const ghalf* p = base + (l & 15) * ld + ((l >> 4) << 3);
  V16U u;
  u.h[0] = *reinterpret_cast<const gv8h*>(p);
  u.h[1] = *reinterpret_cast<const gv8h*>(p + 16);
  return u.v;
}

// B tile (32 K x 16 N) = W^T chunk, loaded from row-major W (out x in), stride ld.
// Lane L: N = L&15, holds K = 8*(L>>4) + {0..7} and +16 (matches A layout).
__device__ __forceinline__ v16h load_b_f16(const h16_t* base, int ld) {
  const int l = lane_id();
  const h16_t* p = base + (l & 15) * ld + ((l >> 4) << 3);
  V16U u;
  u.h[0] = *reinterpret_cast<const v8h*>(p);
  u.h[1] = *reinterpret_cast<const v8h*>(p + 16);
  return u.v;
}
__device__ __forceinline__ v16h load_b_f16(const ghalf* base, int ld) {
  const int l = lane_id();
  const ghalf* p = base + (l & 15) * ld + ((l >> 4) << 3);
  V16U u;
  u.h[0] = *reinterpret_cast<const gv8h*>(p);
  u.h[1] = *reinterpret_cast<const gv8h*>(p + 16);
  return u.v;
}

__device__ __forceinline__ v8f wmma16(v16h a, v16h b, v8f c) {
  return __builtin_amdgcn_wmma_f32_16x16x32_f16(false, a, false, b, (short)0, c,
                                                false, false);
}

// C/D layout: lane L owns column N=L&15, rows M = 8*(L>>4)+r for VGPR r.
__device__ __forceinline__ void store_c_f16(h16_t* dst, int ld, v8f c) {
  const int l = lane_id();
  const int n = l & 15, rb = (l >> 4) << 3;
#pragma unroll
  for (int r = 0; r < 8; ++r) dst[(rb + r) * ld + n] = (h16_t)c[r];
}

// ---------------------------------------------------------------------------
// Software-pipelined multi-tile GEMM: acc[j] += A(16 x KC*32) * B_tile(ntile0+j*TS)
// Next A/B operands are issued before the WMMA consuming the current ones.
// ---------------------------------------------------------------------------
template <int KC, int NT, int TS = 1, typename AP, typename BP>
__device__ __forceinline__ void gemm_tiles(v8f* acc, AP Abase, int lda,
                                           BP Wbase, int ldb, int ntile0) {
  v16h a = load_a_f16(Abase, lda);
  v16h b = load_b_f16(Wbase + ntile0 * 16 * ldb, ldb);
#pragma unroll
  for (int kc = 0; kc < KC; ++kc) {
#pragma unroll
    for (int j = 0; j < NT; ++j) {
      v16h an = a, bn = b;
      if (j + 1 < NT) {
        bn = load_b_f16(Wbase + (ntile0 + (j + 1) * TS) * 16 * ldb + kc * 32,
                        ldb);
      } else if (kc + 1 < KC) {
        bn = load_b_f16(Wbase + ntile0 * 16 * ldb + (kc + 1) * 32, ldb);
        an = load_a_f16(Abase + (kc + 1) * 32, lda);
      }
      acc[j] = wmma16(a, b, acc[j]);
      a = an;
      b = bn;
    }
  }
}

// ---------------------------------------------------------------------------
// Problem constants
// ---------------------------------------------------------------------------
#define NN   256
#define SS   1000
#define DD   64
#define HH   256
#define ZZ   128
#define HID  256

// ---------------------------------------------------------------------------
// Weight f32 -> f16 conversion
// ---------------------------------------------------------------------------
__global__ void k_f32_to_f16(const float* __restrict__ src,
                             h16_t* __restrict__ dst, int n) {
  int i = blockIdx.x * blockDim.x + threadIdx.x;
  if (i < n) dst[i] = (h16_t)src[i];
}

// ---------------------------------------------------------------------------
// Persistent single-WG RNN over reversed sequence.
// LDS: h (256x256 f16 = 128KB) + x slab (256x64 f16 = 32KB) = 160KB dynamic.
// 32 waves; wave w -> mtile = w>>1, 8 n-tiles starting at (w&1)*8.
// ---------------------------------------------------------------------------
__global__ void __launch_bounds__(1024)
k_rnn(const float* __restrict__ X, const float* __restrict__ b_ih,
      const float* __restrict__ b_hh, const h16_t* __restrict__ Wih,
      const h16_t* __restrict__ Whh, h16_t* __restrict__ hlast) {
  extern __shared__ h16_t rnn_sm[];
  h16_t* hbuf = rnn_sm;              // 256*256
  h16_t* xbuf = rnn_sm + HH * HH;    // 256*64
  const int tid = (int)threadIdx.x;
  const int w = tid >> 5;
  const int mtile = w >> 1;
  const int jbase = (w & 1) * 8;
  const int lane = tid & 31;

  for (int e = tid * 8; e < HH * HH; e += 1024 * 8)
    *reinterpret_cast<v8h*>(hbuf + e) = v8h{};

  for (int t = 0; t < SS; ++t) {
    // Xr[:, t] = X[:, S-1-t]  (f32 -> f16 into LDS, float4-wide)
    for (int e = tid * 4; e < NN * DD; e += 1024 * 4) {
      int n = e >> 6, d = e & 63;
      v4f x = *reinterpret_cast<const v4f*>(
          X + (size_t)n * (SS * DD) + (size_t)(SS - 1 - t) * DD + d);
      v4h y;
      y[0] = (h16_t)x[0]; y[1] = (h16_t)x[1];
      y[2] = (h16_t)x[2]; y[3] = (h16_t)x[3];
      *reinterpret_cast<v4h*>(xbuf + e) = y;
    }
    __syncthreads();

    const ghalf* whh = launder_g(Whh);   // force in-loop global weight loads
    const ghalf* wih = launder_g(Wih);

    v8f acc[8] = {};
    gemm_tiles<8, 8>(acc, hbuf + mtile * (16 * HH), HH, whh, HH, jbase);
    gemm_tiles<2, 8>(acc, xbuf + mtile * (16 * DD), DD, wih, DD, jbase);

#pragma unroll
    for (int j = 0; j < 8; ++j) {
      int n = (jbase + j) * 16 + (lane & 15);
      float bias = b_ih[n] + b_hh[n];
#pragma unroll
      for (int r = 0; r < 8; ++r) acc[j][r] = tanhf(acc[j][r] + bias);
    }
    __syncthreads();                   // all reads of old h complete
#pragma unroll
    for (int j = 0; j < 8; ++j)
      store_c_f16(hbuf + mtile * (16 * HH) + (jbase + j) * 16, HH, acc[j]);
    __syncthreads();
  }
  for (int e = tid * 8; e < HH * HH; e += 1024 * 8)
    *reinterpret_cast<v8h*>(hlast + e) = *reinterpret_cast<const v8h*>(hbuf + e);
}

// ---------------------------------------------------------------------------
// Encoder head A: comp = h_last @ W_comp^T + b_comp; z_enc = eps*std + mu (f16)
// 16 blocks x 8 waves; wave (mtile, ztile) computes mu tile (ztile) and std
// tile (ztile+8) of W_comp via tile-stride 8.
// ---------------------------------------------------------------------------
__global__ void __launch_bounds__(256)
k_enc_a(const h16_t* __restrict__ hlast, const h16_t* __restrict__ Wcomp,
        const float* __restrict__ b_comp, const float* __restrict__ eps,
        h16_t* __restrict__ zenc) {
  const int gw = blockIdx.x * 8 + ((int)threadIdx.x >> 5);
  const int mtile = gw >> 3, ztile = gw & 7, lane = (int)threadIdx.x & 31;
  v8f acc[2] = {};
  gemm_tiles<8, 2, 8>(acc, to_g(hlast) + mtile * (16 * HH), HH, to_g(Wcomp),
                      HH, ztile);
  const int z = ztile * 16 + (lane & 15);
  const int rb = (lane >> 4) << 3;
  const float bm = b_comp[z], bs = b_comp[ZZ + z];
#pragma unroll
  for (int r = 0; r < 8; ++r) {
    int m = mtile * 16 + rb + r;
    float mu = acc[0][r] + bm, sd = acc[1][r] + bs;
    zenc[m * ZZ + z] = (h16_t)(eps[m * ZZ + z] * sd + mu);
  }
}

// ---------------------------------------------------------------------------
// Encoder head B: z0 = z_enc @ W_rc^T + b_rc (f32 out)
// ---------------------------------------------------------------------------
__global__ void __launch_bounds__(256)
k_enc_b(const h16_t* __restrict__ zenc, const h16_t* __restrict__ Wrc,
        const float* __restrict__ b_rc, float* __restrict__ z0) {
  const int gw = blockIdx.x * 8 + ((int)threadIdx.x >> 5);
  const int mtile = gw >> 3, ztile = gw & 7, lane = (int)threadIdx.x & 31;
  v8f c[1] = {};
  gemm_tiles<4, 1>(c, to_g(zenc) + mtile * (16 * ZZ), ZZ, to_g(Wrc), ZZ,
                   ztile);
  const int z = ztile * 16 + (lane & 15);
  const int rb = (lane >> 4) << 3;
  const float bb = b_rc[z];
#pragma unroll
  for (int r = 0; r < 8; ++r) z0[(mtile * 16 + rb + r) * ZZ + z] = c[0][r] + bb;
}

// ---------------------------------------------------------------------------
// Fused decoder: out[n, s] = relu(z @ Wd1^T + b1) @ Wd2^T + b2  (per 16-row blk)
// ---------------------------------------------------------------------------
__device__ __forceinline__ void decode_store(
    const h16_t* zin, h16_t* h1, const h16_t* Wd1p, const float* b_d1,
    const h16_t* Wd2p, const float* b_d2, float* out, int nbase, int sidx,
    int w, int lane) {
  const ghalf* Wd1 = launder_g(Wd1p);
  const ghalf* Wd2 = launder_g(Wd2p);
  {  // L1: 16 HID tiles, 2 per wave, K = Z = 128
    v8f c[2] = {};
    gemm_tiles<4, 2>(c, zin, ZZ, Wd1, ZZ, 2 * w);
    const int n0 = (2 * w) * 16 + (lane & 15), n1 = n0 + 16;
    const float bb0 = b_d1[n0], bb1 = b_d1[n1];
#pragma unroll
    for (int r = 0; r < 8; ++r) {
      float x0 = c[0][r] + bb0; c[0][r] = x0 > 0.f ? x0 : 0.f;
      float x1 = c[1][r] + bb1; c[1][r] = x1 > 0.f ? x1 : 0.f;
    }
    store_c_f16(h1 + (2 * w) * 16, HID, c[0]);
    store_c_f16(h1 + (2 * w + 1) * 16, HID, c[1]);
  }
  __syncthreads();
  if (w < 4) {  // L2: 4 D-tiles, K = HID = 256
    v8f c[1] = {};
    gemm_tiles<8, 1>(c, h1, HID, Wd2, HID, w);
    const int d = w * 16 + (lane & 15);
    const int rb = (lane >> 4) << 3;
    const float bb = b_d2[d];
#pragma unroll
    for (int r = 0; r < 8; ++r) {
      int n = nbase + rb + r;
      out[(size_t)n * (SS * DD) + (size_t)sidx * DD + d] = c[0][r] + bb;
    }
  }
  __syncthreads();
}

// ---------------------------------------------------------------------------
// One RK stage: k = f(zin) = Wo3 elu(Wo2 elu(Wo1 zin + b1) + b2) + b3
// ---------------------------------------------------------------------------
__device__ __forceinline__ void f_eval(
    const h16_t* zin, h16_t* h1, h16_t* h2, const h16_t* Wo1p,
    const float* b_o1, const h16_t* Wo2p, const float* b_o2,
    const h16_t* Wo3p, const float* b_o3, float* kbuf, float* accb, float wgt,
    bool first, int w, int lane) {
  const ghalf* Wo1 = launder_g(Wo1p);
  const ghalf* Wo2 = launder_g(Wo2p);
  const ghalf* Wo3 = launder_g(Wo3p);
  {  // L1: Z(128) -> HID(256)
    v8f c[2] = {};
    gemm_tiles<4, 2>(c, zin, ZZ, Wo1, ZZ, 2 * w);
    const int n0 = (2 * w) * 16 + (lane & 15), n1 = n0 + 16;
    const float bb0 = b_o1[n0], bb1 = b_o1[n1];
#pragma unroll
    for (int r = 0; r < 8; ++r) {
      float x0 = c[0][r] + bb0; c[0][r] = x0 > 0.f ? x0 : (expf(x0) - 1.f);
      float x1 = c[1][r] + bb1; c[1][r] = x1 > 0.f ? x1 : (expf(x1) - 1.f);
    }
    store_c_f16(h1 + (2 * w) * 16, HID, c[0]);
    store_c_f16(h1 + (2 * w + 1) * 16, HID, c[1]);
  }
  __syncthreads();
  {  // L2: HID -> HID
    v8f c[2] = {};
    gemm_tiles<8, 2>(c, h1, HID, Wo2, HID, 2 * w);
    const int n0 = (2 * w) * 16 + (lane & 15), n1 = n0 + 16;
    const float bb0 = b_o2[n0], bb1 = b_o2[n1];
#pragma unroll
    for (int r = 0; r < 8; ++r) {
      float x0 = c[0][r] + bb0; c[0][r] = x0 > 0.f ? x0 : (expf(x0) - 1.f);
      float x1 = c[1][r] + bb1; c[1][r] = x1 > 0.f ? x1 : (expf(x1) - 1.f);
    }
    store_c_f16(h2 + (2 * w) * 16, HID, c[0]);
    store_c_f16(h2 + (2 * w + 1) * 16, HID, c[1]);
  }
  __syncthreads();
  {  // L3: HID -> Z, wave w owns z-tile w
    v8f c[1] = {};
    gemm_tiles<8, 1>(c, h2, HID, Wo3, HID, w);
    const int z = w * 16 + (lane & 15);
    const int rb = (lane >> 4) << 3;
    const float bb = b_o3[z];
#pragma unroll
    for (int r = 0; r < 8; ++r) {
      int m = rb + r;
      float kv = c[0][r] + bb;
      kbuf[m * ZZ + z] = kv;
      accb[m * ZZ + z] = first ? kv : (accb[m * ZZ + z] + wgt * kv);
    }
  }
  __syncthreads();
}

// 4-wide elementwise zin = f16(z32 + f*kbuf) over the 16x128 block.
__device__ __forceinline__ void zin_update(h16_t* zin, const float* z32,
                                           const float* kbuf, float f,
                                           int tid) {
#pragma unroll
  for (int e = tid * 4; e < 16 * ZZ; e += 256 * 4) {
    v4f zz = *reinterpret_cast<const v4f*>(z32 + e);
    v4f kk = *reinterpret_cast<const v4f*>(kbuf + e);
    v4h o;
#pragma unroll
    for (int c = 0; c < 4; ++c) o[c] = (h16_t)(zz[c] + f * kk[c]);
    *reinterpret_cast<v4h*>(zin + e) = o;
  }
}

// ---------------------------------------------------------------------------
// Neural-ODE RK4 scan + fused decoder. 16 blocks x 8 waves; block owns 16 rows.
// LDS: z32(8K) kbuf(8K) acc(8K) zin(4K) h1(8K) h2(8K) = 44KB dynamic.
// ---------------------------------------------------------------------------
__global__ void __launch_bounds__(256)
k_ode(const float* __restrict__ tvec, const float* __restrict__ z0,
      const h16_t* __restrict__ Wo1, const float* __restrict__ b_o1,
      const h16_t* __restrict__ Wo2, const float* __restrict__ b_o2,
      const h16_t* __restrict__ Wo3, const float* __restrict__ b_o3,
      const h16_t* __restrict__ Wd1, const float* __restrict__ b_d1,
      const h16_t* __restrict__ Wd2, const float* __restrict__ b_d2,
      float* __restrict__ out) {
  extern __shared__ char ode_sm[];
  float* z32  = (float*)ode_sm;           // 16*128
  float* kbuf = z32 + 16 * ZZ;            // 16*128
  float* accb = kbuf + 16 * ZZ;           // 16*128
  h16_t* zin  = (h16_t*)(accb + 16 * ZZ); // 16*128
  h16_t* h1   = zin + 16 * ZZ;            // 16*256
  h16_t* h2   = h1 + 16 * HID;            // 16*256
  const int tid = (int)threadIdx.x;
  const int w = tid >> 5, lane = tid & 31;
  const int nbase = (int)blockIdx.x * 16;

#pragma unroll
  for (int e = tid * 4; e < 16 * ZZ; e += 256 * 4) {
    v4f zz = *reinterpret_cast<const v4f*>(z0 + nbase * ZZ + e);
    *reinterpret_cast<v4f*>(z32 + e) = zz;
    v4h o;
#pragma unroll
    for (int c = 0; c < 4; ++c) o[c] = (h16_t)zz[c];
    *reinterpret_cast<v4h*>(zin + e) = o;
  }
  __syncthreads();

  decode_store(zin, h1, Wd1, b_d1, Wd2, b_d2, out, nbase, 0, w, lane);

  for (int i = 0; i < SS - 1; ++i) {
    const float dt = (tvec[i + 1] - tvec[i]) * 0.5f;  // / NSUB
    for (int sub = 0; sub < 2; ++sub) {
      f_eval(zin, h1, h2, Wo1, b_o1, Wo2, b_o2, Wo3, b_o3, kbuf, accb,
             1.f, true, w, lane);
      zin_update(zin, z32, kbuf, 0.5f * dt, tid);
      __syncthreads();
      f_eval(zin, h1, h2, Wo1, b_o1, Wo2, b_o2, Wo3, b_o3, kbuf, accb,
             2.f, false, w, lane);
      zin_update(zin, z32, kbuf, 0.5f * dt, tid);
      __syncthreads();
      f_eval(zin, h1, h2, Wo1, b_o1, Wo2, b_o2, Wo3, b_o3, kbuf, accb,
             2.f, false, w, lane);
      zin_update(zin, z32, kbuf, dt, tid);
      __syncthreads();
      f_eval(zin, h1, h2, Wo1, b_o1, Wo2, b_o2, Wo3, b_o3, kbuf, accb,
             1.f, false, w, lane);
#pragma unroll
      for (int e = tid * 4; e < 16 * ZZ; e += 256 * 4) {
        v4f zz = *reinterpret_cast<const v4f*>(z32 + e);
        v4f aa = *reinterpret_cast<const v4f*>(accb + e);
        v4h o;
#pragma unroll
        for (int c = 0; c < 4; ++c) {
          float zn = zz[c] + (dt * (1.f / 6.f)) * aa[c];
          zz[c] = zn;
          o[c] = (h16_t)zn;
        }
        *reinterpret_cast<v4f*>(z32 + e) = zz;
        *reinterpret_cast<v4h*>(zin + e) = o;
      }
      __syncthreads();
    }
    decode_store(zin, h1, Wd1, b_d1, Wd2, b_d2, out, nbase, i + 1, w, lane);
  }
}

// ---------------------------------------------------------------------------
// Host launcher
// ---------------------------------------------------------------------------
extern "C" void kernel_launch(void* const* d_in, const int* in_sizes, int n_in,
                              void* d_out, int out_size, void* d_ws,
                              size_t ws_size, hipStream_t stream) {
  (void)in_sizes; (void)n_in; (void)out_size; (void)ws_size;
  const float* X      = (const float*)d_in[0];
  const float* tvec   = (const float*)d_in[1];
  const float* eps    = (const float*)d_in[2];
  const float* W_ih   = (const float*)d_in[3];
  const float* W_hh   = (const float*)d_in[4];
  const float* b_ih   = (const float*)d_in[5];
  const float* b_hh   = (const float*)d_in[6];
  const float* W_comp = (const float*)d_in[7];
  const float* b_comp = (const float*)d_in[8];
  const float* W_rc   = (const float*)d_in[9];
  const float* b_rc   = (const float*)d_in[10];
  const float* W_o1   = (const float*)d_in[11];
  const float* b_o1   = (const float*)d_in[12];
  const float* W_o2   = (const float*)d_in[13];
  const float* b_o2   = (const float*)d_in[14];
  const float* W_o3   = (const float*)d_in[15];
  const float* b_o3   = (const float*)d_in[16];
  const float* W_d1   = (const float*)d_in[17];
  const float* b_d1   = (const float*)d_in[18];
  const float* W_d2   = (const float*)d_in[19];
  const float* b_d2   = (const float*)d_in[20];
  float* out = (float*)d_out;

  char* ws = (char*)d_ws;  // ~1 MB used total
  h16_t* Wih16   = (h16_t*)(ws + 0);        // 256*64
  h16_t* Whh16   = (h16_t*)(ws + 32768);    // 256*256
  h16_t* Wcomp16 = (h16_t*)(ws + 163840);   // 256*256
  h16_t* Wrc16   = (h16_t*)(ws + 294912);   // 128*128
  h16_t* Wo1_16  = (h16_t*)(ws + 327680);   // 256*128
  h16_t* Wo2_16  = (h16_t*)(ws + 393216);   // 256*256
  h16_t* Wo3_16  = (h16_t*)(ws + 524288);   // 128*256
  h16_t* Wd1_16  = (h16_t*)(ws + 589824);   // 256*128
  h16_t* Wd2_16  = (h16_t*)(ws + 655360);   // 64*256
  h16_t* hlast16 = (h16_t*)(ws + 688128);   // 256*256
  h16_t* zenc16  = (h16_t*)(ws + 819200);   // 256*128
  float* z0f     = (float*)(ws + 884736);   // 256*128 f32

  auto cvt = [&](const float* s, h16_t* d, int n) {
    k_f32_to_f16<<<(n + 255) / 256, 256, 0, stream>>>(s, d, n);
  };
  cvt(W_ih, Wih16, HH * DD);
  cvt(W_hh, Whh16, HH * HH);
  cvt(W_comp, Wcomp16, 2 * ZZ * HH);
  cvt(W_rc, Wrc16, ZZ * ZZ);
  cvt(W_o1, Wo1_16, HID * ZZ);
  cvt(W_o2, Wo2_16, HID * HID);
  cvt(W_o3, Wo3_16, ZZ * HID);
  cvt(W_d1, Wd1_16, HID * ZZ);
  cvt(W_d2, Wd2_16, DD * HID);

  // 160KB dynamic LDS (h 128K + x slab 32K), single persistent workgroup
  k_rnn<<<1, 1024, (HH * HH + NN * DD) * sizeof(h16_t), stream>>>(
      X, b_ih, b_hh, Wih16, Whh16, hlast16);

  k_enc_a<<<16, 256, 0, stream>>>(hlast16, Wcomp16, b_comp, eps, zenc16);
  k_enc_b<<<16, 256, 0, stream>>>(zenc16, Wrc16, b_rc, z0f);

  const size_t ode_sh = 3 * (16 * ZZ) * sizeof(float) +
                        (16 * ZZ + 2 * 16 * HID) * sizeof(h16_t);  // 44 KB
  k_ode<<<16, 256, ode_sh, stream>>>(tvec, z0f, Wo1_16, b_o1, Wo2_16, b_o2,
                                     Wo3_16, b_o3, Wd1_16, b_d1, Wd2_16, b_d2,
                                     out);
}
